// LSTMSimulator_41515153883231
// MI455X (gfx1250) — compile-verified, matrix operand order checked by reading, not run on hardware
//
#include <hip/hip_runtime.h>
#include <hip/hip_bf16.h>

#define H1 64
#define H2 32
#define G1 256   // 4*H1
#define G2 128   // 4*H2
#define BATCH 1024
#define TLEN 8192
#define MT 16        // batch rows per workgroup (WMMA M)
#define NTHREADS 256 // 8 waves

typedef __attribute__((ext_vector_type(16))) _Float16 v16h;
typedef __attribute__((ext_vector_type(8)))  _Float16 v8h;
typedef __attribute__((ext_vector_type(8)))  float    v8f;

#define WMMA_F16(a, b, c) \
    __builtin_amdgcn_wmma_f32_16x16x32_f16(false, (a), false, (b), (short)0, (c), false, false)

// Fast activations: v_exp_f32 + v_rcp_f32 are TRANS32 ops that co-execute
// with regular VALU on CDNA5 — no IEEE divide sequences.
__device__ __forceinline__ float fsig(float x) {
    float e = __builtin_amdgcn_exp2f(x * -1.442695041f);   // e^-x
    return __builtin_amdgcn_rcpf(1.0f + e);
}
__device__ __forceinline__ float ftanh(float x) {
    float e = __builtin_amdgcn_exp2f(x * 2.885390082f);    // e^(2x)
    float r = __builtin_amdgcn_rcpf(1.0f + e);
    return __builtin_fmaf(-2.0f, r, 1.0f);                 // 1 - 2/(e^2x+1)
}

// LDS layout. Weight matrices staged once as per-(ntile,ktile) WMMA B-operand
// tiles (32 lanes x 16 f16, lane-contiguous), then hoisted into VGPRs.
struct SmemLayout {
    _Float16 wB1 [(G1/16) * (H1/32) * 32 * 16]; // 16384 halfs = 32 KB
    _Float16 wB2i[(G2/16) * (H1/32) * 32 * 16]; //  8192 halfs = 16 KB
    _Float16 wB2h[(G2/16) * (H2/32) * 32 * 16]; //  4096 halfs =  8 KB
    _Float16 h1[MT * H1];   // f16 hidden state, row-major (A operand source)
    _Float16 h2[MT * H2];
    float    c1[MT * H1];
    float    c2[MT * H2];
    float    g1[MT * G1];   // layer-1 gate buffer (WMMA D, f32)
    float    g2[MT * G2];   // layer-2 gate buffer (WMMA D, f32)
    float    b1[G1];
    float    b2[G2];
    float    wih1[G1];      // W_ih1 is (256,1) -> 256 floats
    float    wlin[H2 + 1];  // W_lin + b_lin
    float    xb[MT];
};

// Convert an (N,K) row-major f32 weight (used as B = W^T, K x N) into
// per-lane-contiguous f16 WMMA B tiles. 16-bit operand layout: lane half h
// covers k-local {h*8+j} and {16+h*8+j}.
__device__ __forceinline__ void fill_btiles(_Float16* dst, const float* W,
                                            int N, int K) {
    const int KT = K >> 5;
    const int total = N * K;
    for (int idx = threadIdx.x; idx < total; idx += NTHREADS) {
        int j    = idx & 15;
        int lane = (idx >> 4) & 31;
        int tile = idx >> 9;          // 512 halfs per tile
        int kt   = tile % KT;
        int nt   = tile / KT;
        int n    = nt * 16 + (lane & 15);
        int h    = lane >> 4;
        int kl   = (j < 8) ? (h * 8 + j) : (16 + h * 8 + (j - 8));
        int k    = kt * 32 + kl;
        dst[idx] = (_Float16)W[n * K + k];
    }
}

// Assemble this lane's v16h A fragment from row-major f16 state in LDS.
__device__ __forceinline__ v16h load_a(const _Float16* hbuf, int K, int m,
                                       int kt, int hi) {
    const _Float16* p = hbuf + m * K + kt * 32 + hi * 8;
    v8h lo = *(const v8h*)(p);
    v8h hh = *(const v8h*)(p + 16);
    v16h a;
#pragma unroll
    for (int j = 0; j < 8; ++j) { a[j] = lo[j]; a[8 + j] = hh[j]; }
    return a;
}

__global__ __launch_bounds__(NTHREADS) void lstm2_wmma_kernel(
    const float* __restrict__ input,
    const float* __restrict__ W_ih1, const float* __restrict__ W_hh1,
    const float* __restrict__ b_ih1, const float* __restrict__ b_hh1,
    const float* __restrict__ W_ih2, const float* __restrict__ W_hh2,
    const float* __restrict__ b_ih2, const float* __restrict__ b_hh2,
    const float* __restrict__ W_lin, const float* __restrict__ b_lin,
    float* __restrict__ out) {
    __shared__ SmemLayout s;
    const int tid  = threadIdx.x;
    const int wave = tid >> 5;
    const int lane = tid & 31;
    const int mrow = lane & 15; // A row / C column index for this lane
    const int hi   = lane >> 4; // lane half
    const int m0   = blockIdx.x * MT;

    // ---------------- one-time setup ----------------
    fill_btiles(s.wB1,  W_hh1, G1, H1);
    fill_btiles(s.wB2i, W_ih2, G2, H1);
    fill_btiles(s.wB2h, W_hh2, G2, H2);
    for (int i = tid; i < G1; i += NTHREADS) {
        s.b1[i] = b_ih1[i] + b_hh1[i];
        s.wih1[i] = W_ih1[i];
    }
    for (int i = tid; i < G2; i += NTHREADS) s.b2[i] = b_ih2[i] + b_hh2[i];
    if (tid < H2) s.wlin[tid] = W_lin[tid];
    if (tid == H2) s.wlin[H2] = b_lin[0];
    for (int i = tid; i < MT * H1; i += NTHREADS) {
        s.c1[i] = 0.f; s.h1[i] = (_Float16)0.f;
    }
    for (int i = tid; i < MT * H2; i += NTHREADS) {
        s.c2[i] = 0.f; s.h2[i] = (_Float16)0.f;
    }
    __syncthreads();

    // ---- hoist all loop-invariant operands into VGPRs (7 x v16h = 56 VGPRs)
    const v16h B1_00 = *(const v16h*)(s.wB1  + ((( wave      * 2 + 0) * 32 + lane) * 16));
    const v16h B1_01 = *(const v16h*)(s.wB1  + ((( wave      * 2 + 1) * 32 + lane) * 16));
    const v16h B1_10 = *(const v16h*)(s.wB1  + ((((wave + 8) * 2 + 0) * 32 + lane) * 16));
    const v16h B1_11 = *(const v16h*)(s.wB1  + ((((wave + 8) * 2 + 1) * 32 + lane) * 16));
    const v16h B2i_0 = *(const v16h*)(s.wB2i + ((( wave      * 2 + 0) * 32 + lane) * 16));
    const v16h B2i_1 = *(const v16h*)(s.wB2i + ((( wave      * 2 + 1) * 32 + lane) * 16));
    const v16h B2h_0 = *(const v16h*)(s.wB2h + ((  wave                * 32 + lane) * 16));

    // act1: gate column k = (it*256+tid) & 63 == tid & 63 -> thread-invariant
    const int   kq   = tid & (H1 - 1);
    const float wi0  = s.wih1[kq];
    const float wi1  = s.wih1[H1 + kq];
    const float wi2  = s.wih1[2 * H1 + kq];
    const float wi3  = s.wih1[3 * H1 + kq];
    const float bb10 = s.b1[kq];
    const float bb11 = s.b1[H1 + kq];
    const float bb12 = s.b1[2 * H1 + kq];
    const float bb13 = s.b1[3 * H1 + kq];
    // act2: k = tid & 31 -> thread-invariant
    const int   k2q  = tid & (H2 - 1);
    const float bb20 = s.b2[k2q];
    const float bb21 = s.b2[H2 + k2q];
    const float bb22 = s.b2[2 * H2 + k2q];
    const float bb23 = s.b2[3 * H2 + k2q];

    // strided base pointers for the act phases (constant ds offsets after unroll)
    const float*    g1base = &s.g1[(tid >> 6) * G1 + kq];   // stride 4*G1 per it
    float*          c1base = &s.c1[tid];                    // stride 256 per it
    _Float16*       h1base = &s.h1[tid];
    const float*    xbase  = &s.xb[tid >> 6];               // stride 4 per it
    const float*    g2base = &s.g2[(tid >> 5) * G2 + k2q];  // stride 8*G2 per it
    float*          c2base = &s.c2[tid];                    // stride 256 per it
    _Float16*       h2base = &s.h2[tid];

    // Phase A: 7 WMMAs -> g2(t) [if do_g2] and g1(t+1). a0/a1 = h1(t) shared.
    // All accumulator chains start from the inline-0 C operand; biases are
    // applied in the act phases from registers.
    auto phaseA = [&](bool do_g2) {
        v16h a0 = load_a(s.h1, H1, mrow, 0, hi);
        v16h a1 = load_a(s.h1, H1, mrow, 1, hi);
        if (do_g2) {
            v16h a2 = load_a(s.h2, H2, mrow, 0, hi);
            v8f z2{};
            v8f c2 = WMMA_F16(a0, B2i_0, z2);
            c2 = WMMA_F16(a1, B2i_1, c2);
            c2 = WMMA_F16(a2, B2h_0, c2);
#pragma unroll
            for (int r = 0; r < 8; ++r)
                s.g2[(hi * 8 + r) * G2 + wave * 16 + mrow] = c2[r];
        }
        v8f z0{}, z1{};
        v8f c0 = WMMA_F16(a0, B1_00, z0);
        v8f c1 = WMMA_F16(a0, B1_10, z1);
        c0 = WMMA_F16(a1, B1_01, c0);
        c1 = WMMA_F16(a1, B1_11, c1);
#pragma unroll
        for (int r = 0; r < 8; ++r) {
            s.g1[(hi * 8 + r) * G1 + wave * 16 + mrow]       = c0[r];
            s.g1[(hi * 8 + r) * G1 + (wave + 8) * 16 + mrow] = c1[r];
        }
    };

    auto act1 = [&]() { // h1,c1 <- g1, xb
#pragma unroll
        for (int it = 0; it < (MT * H1) / NTHREADS; ++it) {
            const float* g = g1base + it * 4 * G1;
            float xv = xbase[it * 4];
            float iv = (g[0]      + bb10) + xv * wi0;
            float fv = (g[H1]     + bb11) + xv * wi1;
            float gv = (g[2 * H1] + bb12) + xv * wi2;
            float ov = (g[3 * H1] + bb13) + xv * wi3;
            float cn = fsig(fv) * c1base[it * NTHREADS] + fsig(iv) * ftanh(gv);
            c1base[it * NTHREADS] = cn;
            h1base[it * NTHREADS] = (_Float16)(fsig(ov) * ftanh(cn));
        }
    };

    auto act2 = [&]() { // h2,c2 <- g2
#pragma unroll
        for (int it = 0; it < (MT * H2) / NTHREADS; ++it) {
            const float* g = g2base + it * 8 * G2;
            float iv = g[0]      + bb20;
            float fv = g[H2]     + bb21;
            float gv = g[2 * H2] + bb22;
            float ov = g[3 * H2] + bb23;
            float cn = fsig(fv) * c2base[it * NTHREADS] + fsig(iv) * ftanh(gv);
            c2base[it * NTHREADS] = cn;
            h2base[it * NTHREADS] = (_Float16)(fsig(ov) * ftanh(cn));
        }
    };

    // ---------------- prologue: produce h1(0) ----------------
    if (wave == 0 && lane < MT)
        s.xb[lane] = input[(m0 + lane) * TLEN + 0];
    phaseA(false);          // g1(0) = 0 (+b1 in act1); h1 state is zero
    __syncthreads();
    act1();                 // h1(0), c1(0)
    __syncthreads();

    // ---------------- pipelined loop: 2 barriers per step ----------------
    for (int t = 0; t < TLEN; ++t) {
        // Phase A(t): GEMM2(t) + GEMM1(t+1); side jobs off critical path
        if (wave == 0) {
            if (lane < MT && t + 1 < TLEN) {
                s.xb[lane] = input[(m0 + lane) * TLEN + (t + 1)];
                if (t + 65 < TLEN)
                    __builtin_prefetch(&input[(m0 + lane) * TLEN + t + 65], 0, 0);
            }
        } else if (wave == 4 && lane < MT && t > 0) {
            const v8h* hp = (const v8h*)&s.h2[lane * H2];
            float acc = s.wlin[H2];
#pragma unroll
            for (int v = 0; v < 4; ++v) {
                v8h hv = hp[v];
#pragma unroll
                for (int j = 0; j < 8; ++j)
                    acc += (float)hv[j] * s.wlin[v * 8 + j];
            }
            out[(m0 + lane) * TLEN + (t - 1)] = acc;
        }
        phaseA(true);
        __syncthreads();

        // Phase B(t): act2(t) then act1(t+1)
        act2();
        act1();   // last iteration writes unused h1(TLEN); harmless
        __syncthreads();
    }

    // epilogue: y(TLEN-1)
    if (tid < MT) {
        const v8h* hp = (const v8h*)&s.h2[tid * H2];
        float acc = s.wlin[H2];
#pragma unroll
        for (int v = 0; v < 4; ++v) {
            v8h hv = hp[v];
#pragma unroll
            for (int j = 0; j < 8; ++j)
                acc += (float)hv[j] * s.wlin[v * 8 + j];
        }
        out[(m0 + tid) * TLEN + (TLEN - 1)] = acc;
    }
}

extern "C" void kernel_launch(void* const* d_in, const int* in_sizes, int n_in,
                              void* d_out, int out_size, void* d_ws, size_t ws_size,
                              hipStream_t stream) {
    const float* input = (const float*)d_in[0];
    const float* W_ih1 = (const float*)d_in[1];
    const float* W_hh1 = (const float*)d_in[2];
    const float* b_ih1 = (const float*)d_in[3];
    const float* b_hh1 = (const float*)d_in[4];
    const float* W_ih2 = (const float*)d_in[5];
    const float* W_hh2 = (const float*)d_in[6];
    const float* b_ih2 = (const float*)d_in[7];
    const float* b_hh2 = (const float*)d_in[8];
    const float* W_lin = (const float*)d_in[9];
    const float* b_lin = (const float*)d_in[10];
    float* out = (float*)d_out;

    lstm2_wmma_kernel<<<dim3(BATCH / MT), dim3(NTHREADS), 0, stream>>>(
        input, W_ih1, W_hh1, b_ih1, b_hh1, W_ih2, W_hh2, b_ih2, b_hh2,
        W_lin, b_lin, out);
}
